// SharedSelfAttentionModel_7447473292074
// MI455X (gfx1250) — matrix operand order repaired and checked by reading, not compiled, not run on hardware
//
#include <hip/hip_runtime.h>
#include <hip/hip_bf16.h>

typedef __attribute__((ext_vector_type(16))) __bf16 v16bf;
typedef __attribute__((ext_vector_type(8)))  float  v8f;

#define NTOK   16384
#define DMODEL 512
#define SEQL   1024
#define NHEAD  8
#define HDIM   64
#define NLAYER 4

// ---------------- helpers ----------------

__device__ inline unsigned short f32_to_bf16_bits(float f) {
  unsigned int x = __float_as_uint(f);
  x += 0x7FFFu + ((x >> 16) & 1u);           // round-to-nearest-even
  return (unsigned short)(x >> 16);
}

__device__ inline v8f wmma_bf16(v16bf a, v16bf b, v8f c) {
  return __builtin_amdgcn_wmma_f32_16x16x32_bf16(false, a, false, b, (short)0, c,
                                                 false, false);
}

// A-matrix 16x32 bf16 fragment from LDS, rows row-major with stride lda.
// Lane (half,idx): row = idx; VGPR j holds K pair kb,kb+1 with
// kb = 16*half + 2*j  (contiguous, matching frag_b).
__device__ inline v16bf frag_a(const unsigned short* base, int lda, int idx, int half) {
  union { v16bf v; unsigned int u[8]; } f;
  const unsigned short* rp = base + idx * lda;
  #pragma unroll
  for (int j = 0; j < 8; ++j) {
    int kb = 16 * half + 2 * j;
    f.u[j] = *(const unsigned int*)(rp + kb);
  }
  return f.v;
}

// B-matrix 32x16 bf16 fragment from LDS stored transposed as Bt[n][k] (stride ldb).
// Lane (half,idx): col = idx; elements K = 16*half + 0..15 (contiguous 32B).
__device__ inline v16bf frag_b(const unsigned short* base, int ldb, int idx, int half) {
  union { v16bf v; unsigned int u[8]; } f;
  const unsigned short* cp = base + idx * ldb + 16 * half;
  #pragma unroll
  for (int j = 0; j < 8; ++j) f.u[j] = *(const unsigned int*)(cp + 2 * j);
  return f.v;
}

// ---------------- converts ----------------

// fp32 -> bf16, 4 elements/thread (b128 in, b64 out)
__global__ __launch_bounds__(256) void cvt_bf16_v4(const float* __restrict__ in,
                                                   unsigned short* __restrict__ out,
                                                   int n4) {
  int i = blockIdx.x * 256 + threadIdx.x;
  if (i < n4) {
    float4 v = ((const float4*)in)[i];
    uint2 r;
    r.x = (unsigned)f32_to_bf16_bits(v.x) | ((unsigned)f32_to_bf16_bits(v.y) << 16);
    r.y = (unsigned)f32_to_bf16_bits(v.z) | ((unsigned)f32_to_bf16_bits(v.w) << 16);
    ((uint2*)out)[i] = r;
  }
}

// weight prep: fp32 B[K][N] -> bf16 Bt[N][K] (one-time per launch)
__global__ __launch_bounds__(256) void cvt_transpose(const float* __restrict__ src,
                                                     unsigned short* __restrict__ dst,
                                                     int K, int N) {
  int i = blockIdx.x * 256 + threadIdx.x;    // over (K*N)/4, 4 consecutive n
  int n4 = N >> 2;
  if (i < K * n4) {
    int k = i / n4, n = (i % n4) * 4;
    float4 v = *(const float4*)(src + (size_t)k * N + n);
    dst[(size_t)(n + 0) * K + k] = f32_to_bf16_bits(v.x);
    dst[(size_t)(n + 1) * K + k] = f32_to_bf16_bits(v.y);
    dst[(size_t)(n + 2) * K + k] = f32_to_bf16_bits(v.z);
    dst[(size_t)(n + 3) * K + k] = f32_to_bf16_bits(v.w);
  }
}

// ---------------- GEMM: C[M,N] = A(bf16)[M,K] @ Bt(bf16)[N,K]^T + bias ----------------
// block tile 64x256, 8 waves in 2x4, wave tile 32x64 (2x4 WMMA tiles), K step 32.
// Double-buffered LDS (one barrier per K-step, next-tile global loads overlap WMMA).
// GUARD=true adds N bounds checks (only needed for the final N=16 projection).

template <bool GUARD>
__global__ __launch_bounds__(256) void gemm_bf16(const unsigned short* __restrict__ A,
                                                 const unsigned short* __restrict__ Bt,
                                                 const float* __restrict__ bias,
                                                 float* __restrict__ Cf,
                                                 unsigned short* __restrict__ Cb,
                                                 int K, int Nsize, int relu) {
  __shared__ unsigned short As[2][64 * 32];    // [m][k]
  __shared__ unsigned short Bs[2][256 * 32];   // [n][k]
  const int tid  = threadIdx.x;
  const int wid  = tid >> 5;
  const int lane = tid & 31;
  const int half = lane >> 4;
  const int idx  = lane & 15;
  const int wm = wid >> 2;      // 0..1
  const int wn = wid & 3;       // 0..3
  const int m0 = blockIdx.x * 64;
  const int n0 = blockIdx.y * 256;

  v8f acc[2][4];
  #pragma unroll
  for (int mt = 0; mt < 2; ++mt)
    #pragma unroll
    for (int nt = 0; nt < 4; ++nt) acc[mt][nt] = (v8f){};

  // per-thread staging coordinates (all b128 chunks)
  const int ar = tid >> 2, ac = (tid & 3) * 8;             // A: 1 chunk/thread
  const unsigned short* Arow = A + (size_t)(m0 + ar) * K + ac;
  const unsigned short* Brow[4];
  bool bok[4];
  #pragma unroll
  for (int it = 0; it < 4; ++it) {                         // B: 4 chunks/thread
    int i = tid + it * 256;
    int bn = i >> 2, bc = (i & 3) * 8;
    int gn = n0 + bn;
    bok[it] = !GUARD || (gn < Nsize);
    int gnc = GUARD ? (gn < Nsize ? gn : Nsize - 1) : gn;
    Brow[it] = Bt + (size_t)gnc * K + bc;
  }

  // prologue: stage tile 0
  uint4 areg = *(const uint4*)(Arow);
  uint4 breg[4];
  #pragma unroll
  for (int it = 0; it < 4; ++it) {
    uint4 v; v.x = v.y = v.z = v.w = 0u;
    if (bok[it]) v = *(const uint4*)(Brow[it]);
    breg[it] = v;
  }
  ((uint4*)As[0])[tid] = areg;
  #pragma unroll
  for (int it = 0; it < 4; ++it) ((uint4*)Bs[0])[tid + it * 256] = breg[it];
  __syncthreads();

  const int nsteps = K >> 5;
  for (int s = 0; s < nsteps; ++s) {
    const int cur = s & 1;
    const bool more = (s + 1) < nsteps;
    if (more) {                                            // prefetch next tile
      int k1 = (s + 1) << 5;
      areg = *(const uint4*)(Arow + k1);
      #pragma unroll
      for (int it = 0; it < 4; ++it) {
        uint4 v; v.x = v.y = v.z = v.w = 0u;
        if (bok[it]) v = *(const uint4*)(Brow[it] + k1);
        breg[it] = v;
      }
    }

    const unsigned short* as = As[cur];
    const unsigned short* bs = Bs[cur];
    v16bf a0 = frag_a(as + (wm * 32 + 0) * 32, 32, idx, half);
    v16bf a1 = frag_a(as + (wm * 32 + 16) * 32, 32, idx, half);
    v16bf b0 = frag_b(bs + (wn * 64 + 0) * 32, 32, idx, half);
    v16bf b1 = frag_b(bs + (wn * 64 + 16) * 32, 32, idx, half);
    v16bf b2 = frag_b(bs + (wn * 64 + 32) * 32, 32, idx, half);
    v16bf b3 = frag_b(bs + (wn * 64 + 48) * 32, 32, idx, half);
    acc[0][0] = wmma_bf16(a0, b0, acc[0][0]);
    acc[1][0] = wmma_bf16(a1, b0, acc[1][0]);
    acc[0][1] = wmma_bf16(a0, b1, acc[0][1]);
    acc[1][1] = wmma_bf16(a1, b1, acc[1][1]);
    acc[0][2] = wmma_bf16(a0, b2, acc[0][2]);
    acc[1][2] = wmma_bf16(a1, b2, acc[1][2]);
    acc[0][3] = wmma_bf16(a0, b3, acc[0][3]);
    acc[1][3] = wmma_bf16(a1, b3, acc[1][3]);

    if (more) {                                            // commit next tile
      const int nxt = 1 - cur;
      ((uint4*)As[nxt])[tid] = areg;
      #pragma unroll
      for (int it = 0; it < 4; ++it) ((uint4*)Bs[nxt])[tid + it * 256] = breg[it];
      __syncthreads();
    }
  }

  // epilogue: C-layout element j -> (row = j + 8*half, col = idx)
  #pragma unroll
  for (int mt = 0; mt < 2; ++mt) {
    #pragma unroll
    for (int nt = 0; nt < 4; ++nt) {
      int col = n0 + wn * 64 + nt * 16 + idx;
      if (!GUARD || col < Nsize) {
        float bv = bias[col];
        #pragma unroll
        for (int j = 0; j < 8; ++j) {
          int row = m0 + wm * 32 + mt * 16 + j + 8 * half;
          float v = acc[mt][nt][j] + bv;
          if (relu) v = fmaxf(v, 0.0f);
          size_t o = (size_t)row * Nsize + col;
          if (Cf) Cf[o] = v;
          if (Cb) Cb[o] = f32_to_bf16_bits(v);
        }
      }
    }
  }
}

// ---------------- flash attention with ALiBi ----------------
// grid = (N*H, L/128). 8 waves; each wave owns 16 query rows; key tiles of 64.
// qkv bf16 layout: tok*1536 + {0,512,1024} + head*64 + d.

__global__ __launch_bounds__(256) void attn_kernel(const unsigned short* __restrict__ qkv,
                                                   const float* __restrict__ slopes,
                                                   float* __restrict__ out) {
  __shared__ unsigned short Qs[128 * 64];      // [qrow][d]
  __shared__ unsigned short Ks[64 * 64];       // [key][d]   (B-frag layout for Q@K^T)
  __shared__ unsigned short Vt[64 * 64];       // [d][key]   (B-frag layout for P@V)
  __shared__ unsigned short Ps[8][16 * 64];    // per-wave P tile [row][key]

  const int nb = blockIdx.x;
  const int n  = nb >> 3;
  const int hh = nb & 7;
  const int q0 = blockIdx.y * 128;
  const int tid  = threadIdx.x;
  const int wid  = tid >> 5;
  const int lane = tid & 31;
  const int half = lane >> 4;
  const int idx  = lane & 15;
  const float scale = 0.125f;                  // 1/sqrt(64)
  const float slope = slopes[hh];

  const size_t seq_base = (size_t)n * SEQL * 1536;
  const unsigned short* qg = qkv + seq_base + (size_t)hh * 64;
  const unsigned short* kg = qkv + seq_base + 512 + (size_t)hh * 64;
  const unsigned short* vg = qkv + seq_base + 1024 + (size_t)hh * 64;

  // stage Q tile: 128 rows x 64, b128 copies (4 chunks/thread)
  #pragma unroll
  for (int i = tid; i < 128 * 8; i += 256) {
    int r = i >> 3, ch = (i & 7) * 8;
    ((uint4*)Qs)[i] = *(const uint4*)(qg + (size_t)(q0 + r) * 1536 + ch);
  }
  __syncthreads();

  const v16bf aq0 = frag_a(Qs + (wid * 16) * 64, 64, idx, half);       // d 0..31
  const v16bf aq1 = frag_a(Qs + (wid * 16) * 64 + 32, 64, idx, half);  // d 32..63

  v8f o0 = {}, o1 = {}, o2 = {}, o3 = {};
  float mrow[8], lrow[8];
  #pragma unroll
  for (int j = 0; j < 8; ++j) { mrow[j] = -1e30f; lrow[j] = 0.0f; }

  for (int kt = 0; kt < SEQL / 64; ++kt) {
    __syncthreads();
    // stage K (straight) and V (transposed): 2 chunks/thread each
    #pragma unroll
    for (int i = tid; i < 64 * 8; i += 256) {
      int r = i >> 3, ch = (i & 7) * 8;
      ((uint4*)Ks)[i] = *(const uint4*)(kg + (size_t)(kt * 64 + r) * 1536 + ch);
      union { uint4 q; unsigned short s[8]; } u;
      u.q = *(const uint4*)(vg + (size_t)(kt * 64 + r) * 1536 + ch);
      #pragma unroll
      for (int d2 = 0; d2 < 8; ++d2) Vt[(ch + d2) * 64 + r] = u.s[d2];
    }
    __syncthreads();

    // scores: 16 rows x 64 keys in 4 WMMA C-tiles
    v8f s[4];
    #pragma unroll
    for (int t = 0; t < 4; ++t) {
      v16bf bk0 = frag_b(Ks + (t * 16) * 64, 64, idx, half);
      v16bf bk1 = frag_b(Ks + (t * 16) * 64 + 32, 64, idx, half);
      v8f z = {};
      z = wmma_bf16(aq0, bk0, z);
      z = wmma_bf16(aq1, bk1, z);
      s[t] = z;
    }

    // scale + ALiBi, online softmax (row = (j,half) across 16 idx lanes)
    float mnew[8];
    #pragma unroll
    for (int j = 0; j < 8; ++j) {
      int row = q0 + wid * 16 + j + 8 * half;
      float mx = -1e30f;
      #pragma unroll
      for (int t = 0; t < 4; ++t) {
        int col = kt * 64 + t * 16 + idx;
        float v = s[t][j] * scale - slope * fabsf((float)(row - col));
        s[t][j] = v;
        mx = fmaxf(mx, v);
      }
      #pragma unroll
      for (int off = 1; off < 16; off <<= 1) mx = fmaxf(mx, __shfl_xor(mx, off, 32));
      mnew[j] = fmaxf(mrow[j], mx);
    }
    #pragma unroll
    for (int j = 0; j < 8; ++j) {
      float alpha = __expf(mrow[j] - mnew[j]);
      float rs = 0.0f;
      #pragma unroll
      for (int t = 0; t < 4; ++t) {
        float p = __expf(s[t][j] - mnew[j]);
        s[t][j] = p;
        rs += p;
      }
      #pragma unroll
      for (int off = 1; off < 16; off <<= 1) rs += __shfl_xor(rs, off, 32);
      lrow[j] = lrow[j] * alpha + rs;
      mrow[j] = mnew[j];
      o0[j] *= alpha; o1[j] *= alpha; o2[j] *= alpha; o3[j] *= alpha;
    }

    // spill P (bf16) to per-wave LDS, re-read as A fragments
    unsigned short* pw = Ps[wid];
    #pragma unroll
    for (int t = 0; t < 4; ++t)
      #pragma unroll
      for (int j = 0; j < 8; ++j)
        pw[(j + 8 * half) * 64 + t * 16 + idx] = f32_to_bf16_bits(s[t][j]);

    #pragma unroll
    for (int kf = 0; kf < 2; ++kf) {
      v16bf pa  = frag_a(pw + kf * 32, 64, idx, half);
      v16bf bv0 = frag_b(Vt + (0 * 16) * 64 + kf * 32, 64, idx, half);
      v16bf bv1 = frag_b(Vt + (1 * 16) * 64 + kf * 32, 64, idx, half);
      v16bf bv2 = frag_b(Vt + (2 * 16) * 64 + kf * 32, 64, idx, half);
      v16bf bv3 = frag_b(Vt + (3 * 16) * 64 + kf * 32, 64, idx, half);
      o0 = wmma_bf16(pa, bv0, o0);
      o1 = wmma_bf16(pa, bv1, o1);
      o2 = wmma_bf16(pa, bv2, o2);
      o3 = wmma_bf16(pa, bv3, o3);
    }
  }

  // write normalized output (N, L, H*HD)
  #pragma unroll
  for (int j = 0; j < 8; ++j) {
    int row = q0 + wid * 16 + j + 8 * half;
    float inv = 1.0f / lrow[j];
    size_t base = ((size_t)n * SEQL + row) * DMODEL + (size_t)hh * 64;
    out[base + 0 * 16 + idx] = o0[j] * inv;
    out[base + 1 * 16 + idx] = o1[j] * inv;
    out[base + 2 * 16 + idx] = o2[j] * inv;
    out[base + 3 * 16 + idx] = o3[j] * inv;
  }
}

// ---------------- residual add + LayerNorm (writes fp32 h and bf16 h_bf) ----------------

__global__ __launch_bounds__(256) void add_ln_kernel(float* __restrict__ h,
                                                     unsigned short* __restrict__ hb,
                                                     const float* __restrict__ add,
                                                     const float* __restrict__ g,
                                                     const float* __restrict__ b) {
  __shared__ float rs[8], rq[8];
  const int t = blockIdx.x;               // token
  const int tid = threadIdx.x;
  const size_t base = (size_t)t * DMODEL;
  float v0 = h[base + tid]       + add[base + tid];
  float v1 = h[base + tid + 256] + add[base + tid + 256];
  float s = v0 + v1;
  float q = v0 * v0 + v1 * v1;
  #pragma unroll
  for (int off = 1; off < 32; off <<= 1) {
    s += __shfl_xor(s, off, 32);
    q += __shfl_xor(q, off, 32);
  }
  if ((tid & 31) == 0) { rs[tid >> 5] = s; rq[tid >> 5] = q; }
  __syncthreads();
  float ts = 0.0f, tq = 0.0f;
  #pragma unroll
  for (int w = 0; w < 8; ++w) { ts += rs[w]; tq += rq[w]; }
  const float mu = ts * (1.0f / DMODEL);
  const float var = tq * (1.0f / DMODEL) - mu * mu;
  const float rstd = rsqrtf(var + 1e-5f);
  float y0 = (v0 - mu) * rstd * g[tid]       + b[tid];
  float y1 = (v1 - mu) * rstd * g[tid + 256] + b[tid + 256];
  h[base + tid]        = y0;
  h[base + tid + 256]  = y1;
  hb[base + tid]       = f32_to_bf16_bits(y0);
  hb[base + tid + 256] = f32_to_bf16_bits(y1);
}

// ---------------- orchestration ----------------

extern "C" void kernel_launch(void* const* d_in, const int* in_sizes, int n_in,
                              void* d_out, int out_size, void* d_ws, size_t ws_size,
                              hipStream_t stream) {
  const float* x      = (const float*)d_in[0];
  const float* Wp     = (const float*)d_in[1];
  const float* bp     = (const float*)d_in[2];
  const float* Wqkv   = (const float*)d_in[3];
  const float* bqkv   = (const float*)d_in[4];
  const float* ln1_g  = (const float*)d_in[5];
  const float* ln1_b  = (const float*)d_in[6];
  const float* W1     = (const float*)d_in[7];
  const float* b1     = (const float*)d_in[8];
  const float* W2     = (const float*)d_in[9];
  const float* b2     = (const float*)d_in[10];
  const float* ln2_g  = (const float*)d_in[11];
  const float* ln2_b  = (const float*)d_in[12];
  const float* Wo     = (const float*)d_in[13];
  const float* bo     = (const float*)d_in[14];
  const float* slopes = (const float*)d_in[15];

  size_t off = 0;
  auto carve = [&](size_t bytes) {
    void* p = (char*)d_ws + off;
    off += (bytes + 255) & ~(size_t)255;
    return p;
  };
  unsigned short* x_bf   = (unsigned short*)carve((size_t)NTOK * 64 * 2);
  unsigned short* h_bf   = (unsigned short*)carve((size_t)NTOK * DMODEL * 2);
  unsigned short* qkv_bf = (unsigned short*)carve((size_t)NTOK * 1536 * 2);
  unsigned short* ffn_bf = (unsigned short*)carve((size_t)NTOK * 2048 * 2);
  float*          h      = (float*)carve((size_t)NTOK * DMODEL * 4);
  float*          tmp    = (float*)carve((size_t)NTOK * DMODEL * 4);
  unsigned short* wp_t   = (unsigned short*)carve((size_t)512 * 64 * 2);
  unsigned short* wqkv_t = (unsigned short*)carve((size_t)NLAYER * 1536 * 512 * 2);
  unsigned short* w1_t   = (unsigned short*)carve((size_t)NLAYER * 2048 * 512 * 2);
  unsigned short* w2_t   = (unsigned short*)carve((size_t)NLAYER * 512 * 2048 * 2);
  unsigned short* wo_t   = (unsigned short*)carve((size_t)16 * 512 * 2);
  (void)ws_size; (void)n_in; (void)in_sizes; (void)out_size;

  // ---- one-time weight prep: bf16 + transpose to [n][k] ----
  auto tgrid = [](int K, int N) { return dim3((K * (N >> 2) + 255) / 256); };
  cvt_transpose<<<tgrid(64, 512), 256, 0, stream>>>(Wp, wp_t, 64, 512);
  for (int l = 0; l < NLAYER; ++l) {
    cvt_transpose<<<tgrid(512, 1536), 256, 0, stream>>>(
        Wqkv + (size_t)l * 512 * 1536, wqkv_t + (size_t)l * 1536 * 512, 512, 1536);
    cvt_transpose<<<tgrid(512, 2048), 256, 0, stream>>>(
        W1 + (size_t)l * 512 * 2048, w1_t + (size_t)l * 2048 * 512, 512, 2048);
    cvt_transpose<<<tgrid(2048, 512), 256, 0, stream>>>(
        W2 + (size_t)l * 2048 * 512, w2_t + (size_t)l * 512 * 2048, 2048, 512);
  }
  cvt_transpose<<<tgrid(512, 16), 256, 0, stream>>>(Wo, wo_t, 512, 16);

  // ---- activations ----
  cvt_bf16_v4<<<(NTOK * 64 / 4 + 255) / 256, 256, 0, stream>>>(x, x_bf, NTOK * 64 / 4);

  // token projection: (16384 x 64) @ (64 x 512) -> h (fp32) + h_bf
  gemm_bf16<false><<<dim3(NTOK / 64, 2), 256, 0, stream>>>(
      x_bf, wp_t, bp, h, h_bf, 64, DMODEL, 0);

  for (int l = 0; l < NLAYER; ++l) {
    // qkv (bf16 only, feeds attention WMMA directly)
    gemm_bf16<false><<<dim3(NTOK / 64, 6), 256, 0, stream>>>(
        h_bf, wqkv_t + (size_t)l * 1536 * 512, bqkv + (size_t)l * 1536,
        nullptr, qkv_bf, DMODEL, 1536, 0);

    attn_kernel<<<dim3(16 * NHEAD, SEQL / 128), 256, 0, stream>>>(qkv_bf, slopes, tmp);

    add_ln_kernel<<<NTOK, 256, 0, stream>>>(h, h_bf, tmp,
        ln1_g + (size_t)l * DMODEL, ln1_b + (size_t)l * DMODEL);

    // FFN
    gemm_bf16<false><<<dim3(NTOK / 64, 8), 256, 0, stream>>>(
        h_bf, w1_t + (size_t)l * 2048 * 512, b1 + (size_t)l * 2048,
        nullptr, ffn_bf, DMODEL, 2048, 1);
    gemm_bf16<false><<<dim3(NTOK / 64, 2), 256, 0, stream>>>(
        ffn_bf, w2_t + (size_t)l * 512 * 2048, b2 + (size_t)l * DMODEL,
        tmp, nullptr, 2048, DMODEL, 0);

    add_ln_kernel<<<NTOK, 256, 0, stream>>>(h, h_bf, tmp,
        ln2_g + (size_t)l * DMODEL, ln2_b + (size_t)l * DMODEL);
  }

  // final projection: (16384 x 512) @ (512 x 16) -> d_out (fp32)
  gemm_bf16<true><<<dim3(NTOK / 64, 1), 256, 0, stream>>>(
      h_bf, wo_t, bo, (float*)d_out, nullptr, DMODEL, 16, 0);
}
